// HunyuanAttention_10763188044206
// MI455X (gfx1250) — compile-verified, matrix operand order checked by reading, not run on hardware
//
#include <hip/hip_runtime.h>
#include <hip/hip_bf16.h>
#include <math.h>

// ---------------- problem constants ----------------
constexpr int SEQLEN  = 2048;
constexpr int HIDDEN  = 4096;
constexpr int NHEADS  = 32;
constexpr int NKV     = 8;
constexpr int HD      = 128;
constexpr int QSIZE   = NHEADS * HD;          // 4096
constexpr int KVSIZE  = NKV * HD;             // 1024
constexpr int QKVOUT  = QSIZE + 2 * KVSIZE;   // 6144

// ---------------- WMMA fragment types ----------------
typedef __bf16 bf16_t;
typedef bf16_t v16bf __attribute__((ext_vector_type(16)));
typedef float  v8f   __attribute__((ext_vector_type(8)));
typedef unsigned short v8u16 __attribute__((ext_vector_type(8)));

union FragAB {
    v16bf v;
    v8u16 h[2];
};

__device__ __forceinline__ v8f wmma_bf16(const FragAB& a, const FragAB& b, v8f c) {
    // D = A(16x32 bf16) x B(32x16 bf16) + C(16x16 f32)
    return __builtin_amdgcn_wmma_f32_16x16x32_bf16(
        /*neg_a=*/false, a.v, /*neg_b=*/false, b.v,
        /*c_mod=*/(short)0, c, /*reuse_a=*/false, /*reuse_b=*/false);
}

__device__ __forceinline__ unsigned short f2bf_rne(float x) {
    unsigned int u = __float_as_uint(x);
    u += 0x7FFFu + ((u >> 16) & 1u);   // round-to-nearest-even
    return (unsigned short)(u >> 16);
}

// ---------------- conversion kernels ----------------
__global__ void cvt_bf16(const float* __restrict__ src, unsigned short* __restrict__ dst, int n) {
    int i = blockIdx.x * 256 + threadIdx.x;
    if (i < n) dst[i] = f2bf_rne(src[i]);
}

// src [rows, cols] fp32 -> dst [cols, rows] bf16 ; LDS-tiled for coalesced 16-bit writes.
// rows, cols must be multiples of 32 (4096/6144 here). 256 threads = 32x8.
__global__ __launch_bounds__(256)
void cvt_bf16_transpose(const float* __restrict__ src, unsigned short* __restrict__ dst,
                        int rows, int cols) {
    __shared__ unsigned short tile[32][33];   // +1 pad to dodge bank conflicts
    const int tx = threadIdx.x & 31;
    const int ty = threadIdx.x >> 5;          // 0..7
    const int c0 = blockIdx.x * 32;
    const int r0 = blockIdx.y * 32;
#pragma unroll
    for (int i = 0; i < 32; i += 8)
        tile[ty + i][tx] = f2bf_rne(src[(size_t)(r0 + ty + i) * cols + (c0 + tx)]);
    __syncthreads();
#pragma unroll
    for (int i = 0; i < 32; i += 8)
        dst[(size_t)(c0 + ty + i) * rows + (r0 + tx)] = tile[tx][ty + i];
}

// ---------------- bf16 WMMA GEMM: C[M,N] = A[M,K] * Bt[N,K]^T ----------------
// block tile 64(M) x 256(N), 8 waves as 2(M) x 4(N); wave = 32 rows x 64 cols.
// Per k-step of 32: 12 b128 loads feed 8 WMMAs (B fragments reused across 2 A tiles).
__global__ __launch_bounds__(256)
void gemm_bf16_wmma(const unsigned short* __restrict__ A,
                    const unsigned short* __restrict__ Bt,
                    float* __restrict__ C, int M, int N, int K) {
    const int lane = threadIdx.x & 31;
    const int wave = threadIdx.x >> 5;
    const int hl   = lane >> 4;       // half-wave select (K +0 / +8 chunk)
    const int lrow = lane & 15;

    const int m0 = blockIdx.x * 64  + (wave & 1) * 32;
    const int n0 = blockIdx.y * 256 + (wave >> 1) * 64;

    v8f acc[2][4] = {};
    const unsigned short* arow0 = A + (size_t)(m0 + lrow) * K;
    const unsigned short* arow1 = A + (size_t)(m0 + 16 + lrow) * K;
    const unsigned short* brow[4];
#pragma unroll
    for (int j = 0; j < 4; ++j)
        brow[j] = Bt + (size_t)(n0 + j * 16 + lrow) * K;

    for (int k = 0; k < K; k += 32) {
        // prefetch ~8 iterations ahead (global_prefetch_b8); harmless past row end
        __builtin_prefetch(arow0 + k + 256, 0, 1);
        __builtin_prefetch(arow1 + k + 256, 0, 1);

        FragAB a0, a1;
        a0.h[0] = *(const v8u16*)(arow0 + k + hl * 8);
        a0.h[1] = *(const v8u16*)(arow0 + k + 16 + hl * 8);
        a1.h[0] = *(const v8u16*)(arow1 + k + hl * 8);
        a1.h[1] = *(const v8u16*)(arow1 + k + 16 + hl * 8);
#pragma unroll
        for (int j = 0; j < 4; ++j) {
            __builtin_prefetch(brow[j] + k + 256, 0, 1);
            FragAB b;
            b.h[0] = *(const v8u16*)(brow[j] + k + hl * 8);
            b.h[1] = *(const v8u16*)(brow[j] + k + 16 + hl * 8);
            acc[0][j] = wmma_bf16(a0, b, acc[0][j]);
            acc[1][j] = wmma_bf16(a1, b, acc[1][j]);
        }
    }
    // C-fragment layout: vgpr r -> row + r + hl*8, lane -> col n0 + j*16 + lrow
#pragma unroll
    for (int i = 0; i < 2; ++i)
#pragma unroll
        for (int j = 0; j < 4; ++j)
#pragma unroll
            for (int r = 0; r < 8; ++r)
                C[(size_t)(m0 + i * 16 + r + hl * 8) * N + (n0 + j * 16 + lrow)] = acc[i][j][r];
}

// ---------------- RoPE + pack Q/K (bf16, head-major); scale folded into Q ----------------
__global__ void rope_pack(const float* __restrict__ qkv, const int* __restrict__ positions,
                          unsigned short* __restrict__ Qbf, unsigned short* __restrict__ Kbf) {
    int i = blockIdx.x * 256 + threadIdx.x;           // SEQLEN * (NHEADS+NKV) * 64
    const int total = SEQLEN * (NHEADS + NKV) * 64;
    if (i >= total) return;
    int t = i / ((NHEADS + NKV) * 64);
    int p = i - t * ((NHEADS + NKV) * 64);
    int h = p >> 6;                                   // 0..39
    int d = p & 63;                                   // rotation pair index
    float inv_freq = __powf(10000.0f, -(float)(2 * d) / 128.0f);
    float ang = (float)positions[t] * inv_freq;
    float s, c;
    __sincosf(ang, &s, &c);
    if (h < NHEADS) {
        const float scale = 0.08838834764831845f;     // 1/sqrt(128)
        const float* src = qkv + (size_t)t * QKVOUT + h * HD;
        float x1 = src[d], x2 = src[d + 64];
        unsigned short* dst = Qbf + ((size_t)h * SEQLEN + t) * HD;
        dst[d]      = f2bf_rne((x1 * c - x2 * s) * scale);
        dst[d + 64] = f2bf_rne((x2 * c + x1 * s) * scale);
    } else {
        int kh = h - NHEADS;
        const float* src = qkv + (size_t)t * QKVOUT + QSIZE + kh * HD;
        float x1 = src[d], x2 = src[d + 64];
        unsigned short* dst = Kbf + ((size_t)kh * SEQLEN + t) * HD;
        dst[d]      = f2bf_rne(x1 * c - x2 * s);
        dst[d + 64] = f2bf_rne(x2 * c + x1 * s);
    }
}

// V -> bf16 transposed [kvh][d][t]; indexed with t fastest so writes are contiguous
// (strided fp32 reads served by the 192MB L2 after first touch; V slice is only 8MB).
__global__ void v_transpose(const float* __restrict__ qkv, unsigned short* __restrict__ Vt) {
    int i = blockIdx.x * 256 + threadIdx.x;           // NKV * HD * SEQLEN
    if (i >= NKV * HD * SEQLEN) return;
    int t  = i & (SEQLEN - 1);
    int dd = i >> 11;                                 // kvh*HD + d
    int kh = dd >> 7;
    int d  = dd & 127;
    float v = qkv[(size_t)t * QKVOUT + QSIZE + KVSIZE + kh * HD + d];
    Vt[(size_t)i] = f2bf_rne(v);
}

// ---------------- flash attention: 1 wave per (head, 16-row q tile) ----------------
__global__ __launch_bounds__(32)
void flash_attn(const unsigned short* __restrict__ Qbf,
                const unsigned short* __restrict__ Kbf,
                const unsigned short* __restrict__ Vt,
                unsigned short* __restrict__ ctx) {
    __shared__ unsigned short Pld[16 * 32];           // P staged for C->A layout swap
    const int lane = threadIdx.x & 31;
    const int hl   = lane >> 4;
    const int lrow = lane & 15;
    const int head = blockIdx.x >> 7;                 // SEQLEN/16 = 128 q-tiles
    const int qt   = blockIdx.x & 127;
    const int q0   = qt * 16;
    const int kvh  = head >> 2;                       // GQA group of 4

    // Q tile 16x128 as 4 A-fragments (scale already folded in)
    FragAB qa[4];
    const unsigned short* qrow = Qbf + ((size_t)head * SEQLEN + (q0 + lrow)) * HD;
#pragma unroll
    for (int c = 0; c < 4; ++c) {
        qa[c].h[0] = *(const v8u16*)(qrow + c * 32 + hl * 8);
        qa[c].h[1] = *(const v8u16*)(qrow + c * 32 + 16 + hl * 8);
    }

    v8f o[8] = {};                                    // 16x128 fp32 accumulator
    float rmax[8], rsum[8];
#pragma unroll
    for (int r = 0; r < 8; ++r) { rmax[r] = -1e30f; rsum[r] = 0.0f; }

    const int nkb = (q0 + 16 + 31) >> 5;              // 32-key blocks covering causal span
    for (int kb = 0; kb < nkb; ++kb) {
        const int kbase = kb * 32;

        // S = Q * K^T : two 16x16 tiles, K-dim = 128
        v8f s[2] = {};
#pragma unroll
        for (int st = 0; st < 2; ++st) {
            const unsigned short* krow =
                Kbf + ((size_t)kvh * SEQLEN + (kbase + st * 16 + lrow)) * HD;
#pragma unroll
            for (int c = 0; c < 4; ++c) {
                FragAB b;
                b.h[0] = *(const v8u16*)(krow + c * 32 + hl * 8);
                b.h[1] = *(const v8u16*)(krow + c * 32 + 16 + hl * 8);
                s[st] = wmma_bf16(qa[c], b, s[st]);
            }
        }

        // causal mask: lane holds key = kbase + st*16 + lrow, rows q0 + r + hl*8
#pragma unroll
        for (int st = 0; st < 2; ++st) {
            int key = kbase + st * 16 + lrow;
#pragma unroll
            for (int r = 0; r < 8; ++r)
                if (key > q0 + r + hl * 8) s[st][r] = -1e30f;
        }

        // online softmax (row stats reduced across the 16 lanes of each half-wave)
#pragma unroll
        for (int r = 0; r < 8; ++r) {
            float m = fmaxf(s[0][r], s[1][r]);
#pragma unroll
            for (int x = 1; x < 16; x <<= 1) m = fmaxf(m, __shfl_xor(m, x, 32));
            float nm    = fmaxf(rmax[r], m);
            float alpha = __expf(rmax[r] - nm);
            rmax[r] = nm;
            float p0 = __expf(s[0][r] - nm);
            float p1 = __expf(s[1][r] - nm);
            s[0][r] = p0; s[1][r] = p1;
            float ps = p0 + p1;
#pragma unroll
            for (int x = 1; x < 16; x <<= 1) ps += __shfl_xor(ps, x, 32);
            rsum[r] = rsum[r] * alpha + ps;
#pragma unroll
            for (int j = 0; j < 8; ++j) o[j][r] *= alpha;
        }

        // stage P (bf16, 16x32 row-major) to LDS, reload as an A-fragment
#pragma unroll
        for (int st = 0; st < 2; ++st)
#pragma unroll
            for (int r = 0; r < 8; ++r)
                Pld[(r + hl * 8) * 32 + st * 16 + lrow] = f2bf_rne(s[st][r]);
        __syncthreads();   // single wave: s_wait_dscnt (barrier degenerates to NOP)
        FragAB pa;
        pa.h[0] = *(const v8u16*)(&Pld[lrow * 32 + hl * 8]);
        pa.h[1] = *(const v8u16*)(&Pld[lrow * 32 + 16 + hl * 8]);

        // O += P(16x32) * V(32x128)
#pragma unroll
        for (int j = 0; j < 8; ++j) {
            const unsigned short* vrow = Vt + ((size_t)kvh * HD + j * 16 + lrow) * SEQLEN;
            FragAB b;
            b.h[0] = *(const v8u16*)(vrow + kbase + hl * 8);
            b.h[1] = *(const v8u16*)(vrow + kbase + 16 + hl * 8);
            o[j] = wmma_bf16(pa, b, o[j]);
        }
    }

    // finalize: O /= rowsum, write bf16 ctx [t, head*128 + d]
#pragma unroll
    for (int j = 0; j < 8; ++j) {
#pragma unroll
        for (int r = 0; r < 8; ++r) {
            float val = o[j][r] / rsum[r];
            ctx[(size_t)(q0 + r + hl * 8) * QSIZE + head * HD + j * 16 + lrow] = f2bf_rne(val);
        }
    }
}

// ---------------- launch ----------------
extern "C" void kernel_launch(void* const* d_in, const int* in_sizes, int n_in,
                              void* d_out, int out_size, void* d_ws, size_t ws_size,
                              hipStream_t stream) {
    const float* hidden    = (const float*)d_in[0];
    const int*   positions = (const int*)d_in[1];
    const float* w_qkv     = (const float*)d_in[2];
    const float* w_o       = (const float*)d_in[3];
    float* out = (float*)d_out;

    char* ws = (char*)d_ws;
    size_t off = 0;
    auto alloc = [&](size_t bytes) -> void* {
        void* p = ws + off;
        off += (bytes + 255) & ~(size_t)255;
        return p;
    };
    unsigned short* Abf   = (unsigned short*)alloc((size_t)SEQLEN * HIDDEN * 2);
    unsigned short* WqkvT = (unsigned short*)alloc((size_t)QKVOUT * HIDDEN * 2);
    unsigned short* WoT   = (unsigned short*)alloc((size_t)HIDDEN * QSIZE * 2);
    float*          qkv   = (float*)alloc((size_t)SEQLEN * QKVOUT * 4);
    unsigned short* Qbf   = (unsigned short*)alloc((size_t)NHEADS * SEQLEN * HD * 2);
    unsigned short* Kbf   = (unsigned short*)alloc((size_t)NKV * SEQLEN * HD * 2);
    unsigned short* Vt    = (unsigned short*)alloc((size_t)NKV * HD * SEQLEN * 2);
    unsigned short* Ctx   = (unsigned short*)alloc((size_t)SEQLEN * QSIZE * 2);

    cvt_bf16<<<(SEQLEN * HIDDEN + 255) / 256, 256, 0, stream>>>(hidden, Abf, SEQLEN * HIDDEN);
    cvt_bf16_transpose<<<dim3(QKVOUT / 32, HIDDEN / 32), 256, 0, stream>>>(
        w_qkv, WqkvT, HIDDEN, QKVOUT);
    cvt_bf16_transpose<<<dim3(HIDDEN / 32, QSIZE / 32), 256, 0, stream>>>(
        w_o, WoT, QSIZE, HIDDEN);

    gemm_bf16_wmma<<<dim3(SEQLEN / 64, QKVOUT / 256), 256, 0, stream>>>(
        Abf, WqkvT, qkv, SEQLEN, QKVOUT, HIDDEN);

    rope_pack<<<(SEQLEN * (NHEADS + NKV) * 64 + 255) / 256, 256, 0, stream>>>(
        qkv, positions, Qbf, Kbf);
    v_transpose<<<(NKV * HD * SEQLEN + 255) / 256, 256, 0, stream>>>(qkv, Vt);

    flash_attn<<<NHEADS * (SEQLEN / 16), 32, 0, stream>>>(Qbf, Kbf, Vt, Ctx);

    gemm_bf16_wmma<<<dim3(SEQLEN / 64, HIDDEN / 256), 256, 0, stream>>>(
        Ctx, WoT, out, SEQLEN, HIDDEN, QSIZE);
}
